// SlotAttention_28226525069720
// MI455X (gfx1250) — matrix operand order repaired and checked by reading, not compiled, not run on hardware
//
#include <hip/hip_runtime.h>

// ---------------- problem constants ----------------
#define BB 32
#define NN 4096
#define DD 256
#define SS 8
#define HH 512
#define N_ITERS 3
#define LN_EPS 1e-5f
#define EPS 1e-8f
#define SCALE 0.0625f   // 256^-0.5

typedef __bf16 bf16;
typedef __attribute__((ext_vector_type(8)))  __bf16 v8bf;
typedef __attribute__((ext_vector_type(16))) __bf16 v16bf;
typedef __attribute__((ext_vector_type(8)))  float  v8f;
typedef __attribute__((ext_vector_type(4)))  int    v4i;

// run permutation for the 16-bit WMMA A-operand layout:
// within each 32-K block store 8-element runs in order [0,2,1,3] so a lane's
// fragment (h*16 .. h*16+15) is contiguous.
__device__ __forceinline__ int run_perm(int r) { return (r == 1) ? 2 : ((r == 2) ? 1 : r); }

#if defined(__has_builtin)
#if __has_builtin(__builtin_amdgcn_global_load_async_to_lds_b128)
#define HAVE_ASYNC_LDS 1
#endif
#endif

#ifdef HAVE_ASYNC_LDS
typedef __attribute__((address_space(1))) v4i* as1_v4i;
typedef __attribute__((address_space(3))) v4i* as3_v4i;
__device__ __forceinline__ void async_wait0() {
#if __has_builtin(__builtin_amdgcn_s_wait_asynccnt)
  __builtin_amdgcn_s_wait_asynccnt(0);
#else
  asm volatile("s_wait_asynccnt 0" ::: "memory");
#endif
}
#endif

__device__ __forceinline__ float warp_sum(float x) {
#pragma unroll
  for (int o = 16; o > 0; o >>= 1) x += __shfl_xor(x, o, 32);
  return x;
}
__device__ __forceinline__ float warp_max(float x) {
#pragma unroll
  for (int o = 16; o > 0; o >>= 1) x = fmaxf(x, __shfl_xor(x, o, 32));
  return x;
}
__device__ __forceinline__ v8f zero8() {
  v8f z;
#pragma unroll
  for (int i = 0; i < 8; i++) z[i] = 0.f;
  return z;
}

// ---------------- K0: LayerNorm(inputs) -> bf16 (row-major) -----------------
__global__ void sa_ln_in(const float* __restrict__ x,
                         const float* __restrict__ g,
                         const float* __restrict__ b,
                         bf16* __restrict__ xn) {
  int row  = blockIdx.x * 8 + (threadIdx.x >> 5);
  int lane = threadIdx.x & 31;
  const float* xr = x + (size_t)row * DD;
  float v[8];
  float s = 0.f;
#pragma unroll
  for (int i = 0; i < 8; i++) { v[i] = xr[lane + i * 32]; s += v[i]; }
  s = warp_sum(s);
  float mu = s * (1.f / DD);
  float vs = 0.f;
#pragma unroll
  for (int i = 0; i < 8; i++) { float d = v[i] - mu; vs += d * d; }
  vs = warp_sum(vs);
  float inv = rsqrtf(vs * (1.f / DD) + LN_EPS);
  bf16* o = xn + (size_t)row * DD;
#pragma unroll
  for (int i = 0; i < 8; i++) {
    int c = lane + i * 32;
    o[c] = (bf16)(((v[i] - mu) * inv) * g[c] + b[c]);
  }
}

// ------- K0b: f32 weights -> bf16, B-operand packed:  p[((k/32)*256+n)*32 + k%32] -------
__global__ void sa_cvt_packB(const float* __restrict__ w, bf16* __restrict__ o) {
  int i = blockIdx.x * 256 + threadIdx.x;   // 65536
  int k = i >> 8, n = i & 255;
  o[(((size_t)(k >> 5) * DD) + n) * 32 + (k & 31)] = (bf16)w[i];
}

// ---------------- K1: k = xn@Wk (A-packed out), v = xn@Wv (B-packed out) ----------------
// block = 256 thr (8 waves); LDS holds the 16x256 A panel, natural order.
__global__ void sa_gemm_kv(const bf16* __restrict__ xn,
                           const bf16* __restrict__ wkp,
                           const bf16* __restrict__ wvp,
                           bf16* __restrict__ kpk,   // A-packed: [row][256] natural order
                           bf16* __restrict__ vpk) { // B-packed: [b][n/32][d][n%32]
  __shared__ __align__(32) bf16 As[16][DD];
  int tid = threadIdx.x;
  size_t row0 = (size_t)blockIdx.x * 16;

  // stage A panel, natural order: 512 16-byte chunks, 2 per thread
#pragma unroll
  for (int i = 0; i < 2; i++) {
    int id  = tid + i * 256;
    int row = id >> 5, cj = id & 31;
    int blk = cj >> 2, r0 = cj & 3, p = r0;
    const bf16* gs = xn + (row0 + row) * DD + blk * 32 + r0 * 8;
    bf16*       ld = &As[0][0] + row * DD + blk * 32 + p * 8;
#ifdef HAVE_ASYNC_LDS
    __builtin_amdgcn_global_load_async_to_lds_b128(
        (as1_v4i)(v4i*)(void*)const_cast<bf16*>(gs),
        (as3_v4i)(v4i*)(void*)ld, 0, 0);
#else
    *(v8bf*)ld = *(const v8bf*)gs;
#endif
  }
#ifdef HAVE_ASYNC_LDS
  async_wait0();
#endif
  __syncthreads();

  int w = tid >> 5, lane = tid & 31;
  int c = lane & 15, h = lane >> 4;
  const bf16* W = (w < 4) ? wkp : wvp;
  int cbase = w & 3;

  v8f acc[4];
#pragma unroll
  for (int t = 0; t < 4; t++) acc[t] = zero8();

#pragma unroll
  for (int ks = 0; ks < 8; ks++) {
    v16bf a = *(const v16bf*)&As[c][ks * 32 + h * 16];
#pragma unroll
    for (int t = 0; t < 4; t++) {
      int n0 = (cbase + 4 * t) * 16;
      v16bf bm = *(const v16bf*)&W[(((size_t)ks * DD) + n0 + c) * 32 + h * 16];
      acc[t] = __builtin_amdgcn_wmma_f32_16x16x32_bf16(false, a, false, bm,
                                                       (short)0, acc[t], false, false);
    }
  }
#pragma unroll
  for (int t = 0; t < 4; t++) {
    int n0 = (cbase + 4 * t) * 16;
    int d = n0 + c;
    // k: A-packed (row-major, natural columns)
    int dblk = d >> 5, wd = d & 31, pp = wd >> 3;
    int dcol = dblk * 32 + pp * 8 + (wd & 7);
#pragma unroll
    for (int i = 0; i < 8; i++) {
      size_t gr = row0 + i + 8 * h;           // global token row
      if (w < 4) {
        kpk[gr * DD + dcol] = (bf16)acc[t][i];
      } else {
        size_t bb = gr >> 12;                 // batch
        int    n  = (int)(gr & (NN - 1));
        vpk[(((bb * (NN / 32)) + (n >> 5)) * DD + d) * 32 + (n & 31)] = (bf16)acc[t][i];
      }
    }
  }
}

// ---------------- K2: slots init ----------------
__global__ void sa_slots_init(const float* __restrict__ mu,
                              const float* __restrict__ sigma,
                              const float* __restrict__ noise,
                              float* __restrict__ slots) {
  int i = blockIdx.x * 256 + threadIdx.x;
  int d = i & (DD - 1);
  float sg = sigma[d];
  float sp = (sg > 20.f) ? sg : log1pf(__expf(sg));
  slots[i] = mu[d] + (sp + EPS) * noise[i];
}

// ---------------- K3: q = LN(slots)@Wq -> bf16 (row-major) ----------------
__global__ void sa_qproj(const float* __restrict__ slots,
                         const float* __restrict__ g,
                         const float* __restrict__ b,
                         const float* __restrict__ Wq,
                         bf16* __restrict__ qbf) {
  __shared__ float sln[SS][DD];
  int bb = blockIdx.x, tid = threadIdx.x, s = tid >> 5, lane = tid & 31;
  const float* sr = slots + ((size_t)bb * SS + s) * DD;
  float v[8];
  float sum = 0.f;
#pragma unroll
  for (int i = 0; i < 8; i++) { v[i] = sr[lane + i * 32]; sum += v[i]; }
  sum = warp_sum(sum);
  float mu = sum * (1.f / DD);
  float vs = 0.f;
#pragma unroll
  for (int i = 0; i < 8; i++) { float d = v[i] - mu; vs += d * d; }
  vs = warp_sum(vs);
  float inv = rsqrtf(vs * (1.f / DD) + LN_EPS);
#pragma unroll
  for (int i = 0; i < 8; i++) {
    int c = lane + i * 32;
    sln[s][c] = ((v[i] - mu) * inv) * g[c] + b[c];
  }
  __syncthreads();
  int d = tid;
  float acc[SS];
#pragma unroll
  for (int s2 = 0; s2 < SS; s2++) acc[s2] = 0.f;
  for (int e = 0; e < DD; e++) {
    float w = Wq[e * DD + d];
#pragma unroll
    for (int s2 = 0; s2 < SS; s2++) acc[s2] += sln[s2][e] * w;
  }
#pragma unroll
  for (int s2 = 0; s2 < SS; s2++)
    qbf[((size_t)bb * SS + s2) * DD + d] = (bf16)acc[s2];
}

// ---------------- K4: logits_t[b][s][n] = SCALE * k·q  (WMMA) ----------------
// one wave per 16-row n-tile; A = k (A-packed), B = q^T (row-major q)
__global__ void sa_logits(const bf16* __restrict__ kpk,
                          const bf16* __restrict__ qbf,
                          float* __restrict__ logits) {
  int b = blockIdx.x;
  int w = threadIdx.x >> 5, lane = threadIdx.x & 31;
  int n0 = (blockIdx.y * 8 + w) * 16;
  int c = lane & 15, h = lane >> 4;
  const bf16* kb = kpk + (size_t)b * NN * DD;
  const bf16* qb = qbf + (size_t)b * SS * DD;
  int qs = (c < 8) ? c : 7;   // cols 8..15 computed but ignored
  v8f acc = zero8();
#pragma unroll
  for (int ks = 0; ks < 8; ks++) {
    int k0 = ks * 32;
    v16bf a  = *(const v16bf*)&kb[(size_t)(n0 + c) * DD + k0 + h * 16]; // natural row
    v16bf bm = *(const v16bf*)&qb[(size_t)qs * DD + k0 + h * 16];       // natural row
    acc = __builtin_amdgcn_wmma_f32_16x16x32_bf16(false, a, false, bm,
                                                  (short)0, acc, false, false);
  }
  if (c < 8) {
    v8f outv;
#pragma unroll
    for (int i = 0; i < 8; i++) outv[i] = acc[i] * SCALE;
    // rows i -> n = n0 + 8*h + i : contiguous 8-float store
    *(v8f*)&logits[((size_t)b * SS + c) * NN + n0 + 8 * h] = outv;
  }
}

// ---------------- K5: softmax over N -> attn_t[b][s][n] bf16 ----------------
__global__ void sa_softmax(const float* __restrict__ logits, bf16* __restrict__ attn) {
  __shared__ float redm[8];
  __shared__ float reds[8];
  int b = blockIdx.x, s = blockIdx.y;
  int tid = threadIdx.x, w = tid >> 5, lane = tid & 31;
  const float* lp = logits + ((size_t)b * SS + s) * NN;
  float v[16];
  float mx = -3.4e38f;
#pragma unroll
  for (int i = 0; i < 16; i++) {
    v[i] = lp[tid + i * 256];
    mx = fmaxf(mx, v[i]);
  }
  mx = warp_max(mx);
  if (lane == 0) redm[w] = mx;
  __syncthreads();
  if (tid == 0) {
    float t = redm[0];
#pragma unroll
    for (int i = 1; i < 8; i++) t = fmaxf(t, redm[i]);
    redm[0] = t;
  }
  __syncthreads();
  mx = redm[0];
  float sum = 0.f;
#pragma unroll
  for (int i = 0; i < 16; i++) { v[i] = __expf(v[i] - mx); sum += v[i]; }
  sum = warp_sum(sum);
  if (lane == 0) reds[w] = sum;
  __syncthreads();
  if (tid == 0) {
    float t = 0.f;
#pragma unroll
    for (int i = 0; i < 8; i++) t += reds[i];
    reds[0] = t;
  }
  __syncthreads();
  sum = reds[0];
  float inv = 1.f / (sum * (1.f + EPS));
  bf16* ap = attn + ((size_t)b * SS + s) * NN;
#pragma unroll
  for (int i = 0; i < 16; i++) {
    int n = tid + i * 256;
    int wn = n & 31, p = wn >> 3;
    ap[(n & ~31) + p * 8 + (wn & 7)] = (bf16)(v[i] * inv);
  }
}

// ---------------- K6: slots += attn^T @ v  (WMMA, K=4096) ----------------
// grid (B,4) x block 128; A = attn_t (A-packed), B = v (B-packed)
__global__ void sa_updates(const bf16* __restrict__ attn,
                           const bf16* __restrict__ vpk,
                           float* __restrict__ slots) {
  int b = blockIdx.x;
  int w = threadIdx.x >> 5, lane = threadIdx.x & 31;
  int d0 = (blockIdx.y * 4 + w) * 16;
  int c = lane & 15, h = lane >> 4;
  const bf16* ab = attn + (size_t)b * SS * NN;
  const bf16* vb = vpk + (size_t)b * NN * DD;
  v16bf azero;
#pragma unroll
  for (int j = 0; j < 16; j++) azero[j] = (bf16)0.f;
  v8f acc = zero8();
  for (int ks = 0; ks < NN / 32; ks++) {
    int n0 = ks * 32;
    __builtin_prefetch(&vb[(((size_t)ks + 1) * DD + d0 + c) * 32 + h * 16], 0, 3);
    v16bf a = azero;
    if (c < 8) a = *(const v16bf*)&ab[(size_t)c * NN + n0 + h * 16];
    v16bf bm = *(const v16bf*)&vb[(((size_t)ks) * DD + d0 + c) * 32 + h * 16];
    acc = __builtin_amdgcn_wmma_f32_16x16x32_bf16(false, a, false, bm,
                                                  (short)0, acc, false, false);
  }
  if (h == 0) {   // rows 0..7 live in lanes 0..15, VGPR i -> slot i
    float* sp = slots + (size_t)b * SS * DD;
#pragma unroll
    for (int i = 0; i < 8; i++) sp[i * DD + d0 + c] += acc[i];
  }
}

// ---------------- K7: slots += MLP(LN(slots)) ----------------
__global__ void sa_mlp(float* __restrict__ slots,
                       const float* __restrict__ g,
                       const float* __restrict__ b,
                       const float* __restrict__ W1,
                       const float* __restrict__ b1,
                       const float* __restrict__ W2,
                       const float* __restrict__ b2) {
  __shared__ float hln[SS][DD];
  __shared__ float r[SS][HH];
  int bb = blockIdx.x, tid = threadIdx.x, s = tid >> 5, lane = tid & 31;
  const float* sr = slots + ((size_t)bb * SS + s) * DD;
  float v[8];
  float sum = 0.f;
#pragma unroll
  for (int i = 0; i < 8; i++) { v[i] = sr[lane + i * 32]; sum += v[i]; }
  sum = warp_sum(sum);
  float mu = sum * (1.f / DD);
  float vs = 0.f;
#pragma unroll
  for (int i = 0; i < 8; i++) { float d = v[i] - mu; vs += d * d; }
  vs = warp_sum(vs);
  float inv = rsqrtf(vs * (1.f / DD) + LN_EPS);
#pragma unroll
  for (int i = 0; i < 8; i++) {
    int c = lane + i * 32;
    hln[s][c] = ((v[i] - mu) * inv) * g[c] + b[c];
  }
  __syncthreads();
#pragma unroll
  for (int jj = 0; jj < 2; jj++) {
    int j = tid + jj * 256;
    float acc[SS];
#pragma unroll
    for (int s2 = 0; s2 < SS; s2++) acc[s2] = 0.f;
    for (int e = 0; e < DD; e++) {
      float wv = W1[e * HH + j];
#pragma unroll
      for (int s2 = 0; s2 < SS; s2++) acc[s2] += hln[s2][e] * wv;
    }
#pragma unroll
    for (int s2 = 0; s2 < SS; s2++) r[s2][j] = fmaxf(acc[s2] + b1[j], 0.f);
  }
  __syncthreads();
  int d = tid;
  float acc2[SS];
#pragma unroll
  for (int s2 = 0; s2 < SS; s2++) acc2[s2] = 0.f;
  for (int j = 0; j < HH; j++) {
    float wv = W2[j * DD + d];
#pragma unroll
    for (int s2 = 0; s2 < SS; s2++) acc2[s2] += r[s2][j] * wv;
  }
#pragma unroll
  for (int s2 = 0; s2 < SS; s2++)
    slots[((size_t)bb * SS + s2) * DD + d] += acc2[s2] + b2[d];
}

// ---------------- host launcher ----------------
extern "C" void kernel_launch(void* const* d_in, const int* in_sizes, int n_in,
                              void* d_out, int out_size, void* d_ws, size_t ws_size,
                              hipStream_t stream) {
  const float* x       = (const float*)d_in[0];
  const float* noise   = (const float*)d_in[1];
  const float* mu      = (const float*)d_in[2];
  const float* sigma   = (const float*)d_in[3];
  const float* Wq      = (const float*)d_in[4];
  const float* Wk      = (const float*)d_in[5];
  const float* Wv      = (const float*)d_in[6];
  const float* ln_in_g = (const float*)d_in[7];
  const float* ln_in_b = (const float*)d_in[8];
  const float* ln_s_g  = (const float*)d_in[9];
  const float* ln_s_b  = (const float*)d_in[10];
  const float* ln_m_g  = (const float*)d_in[11];
  const float* ln_m_b  = (const float*)d_in[12];
  const float* W1      = (const float*)d_in[13];
  const float* b1      = (const float*)d_in[14];
  const float* W2      = (const float*)d_in[15];
  const float* b2      = (const float*)d_in[16];

  char* ws = (char*)d_ws;
  size_t off = 0;
  auto take = [&](size_t bytes) {
    void* p = ws + off;
    off = (off + bytes + 255) & ~(size_t)255;
    return p;
  };
  const size_t BN = (size_t)BB * NN;
  bf16* xn     = (bf16*)take(BN * DD * sizeof(bf16));
  bf16* kpk    = (bf16*)take(BN * DD * sizeof(bf16));
  bf16* vpk    = (bf16*)take(BN * DD * sizeof(bf16));
  bf16* wkp    = (bf16*)take((size_t)DD * DD * sizeof(bf16));
  bf16* wvp    = (bf16*)take((size_t)DD * DD * sizeof(bf16));
  bf16* qbf    = (bf16*)take((size_t)BB * SS * DD * sizeof(bf16));
  float* logit = (float*)take(BN * SS * sizeof(float));
  bf16* attn   = (bf16*)take(BN * SS * sizeof(bf16));
  (void)ws_size; (void)in_sizes; (void)n_in; (void)out_size;

  float* slots = (float*)d_out;   // slots live in d_out

  sa_ln_in<<<BN / 8, 256, 0, stream>>>(x, ln_in_g, ln_in_b, xn);
  sa_cvt_packB<<<(DD * DD + 255) / 256, 256, 0, stream>>>(Wk, wkp);
  sa_cvt_packB<<<(DD * DD + 255) / 256, 256, 0, stream>>>(Wv, wvp);
  sa_slots_init<<<(BB * SS * DD + 255) / 256, 256, 0, stream>>>(mu, sigma, noise, slots);
  sa_gemm_kv<<<BN / 16, 256, 0, stream>>>(xn, wkp, wvp, kpk, vpk);

  for (int it = 0; it < N_ITERS; it++) {
    sa_qproj<<<BB, 256, 0, stream>>>(slots, ln_s_g, ln_s_b, Wq, qbf);
    sa_logits<<<dim3(BB, NN / 128), 256, 0, stream>>>(kpk, qbf, logit);
    sa_softmax<<<dim3(BB, SS), 256, 0, stream>>>(logit, attn);
    sa_updates<<<dim3(BB, 4), 128, 0, stream>>>(attn, vpk, slots);
    sa_mlp<<<BB, 256, 0, stream>>>(slots, ln_m_g, ln_m_b, W1, b1, W2, b2);
  }
}
